// GNN_LP_Model_40372692582488
// MI455X (gfx1250) — matrix operand order repaired and checked by reading, not compiled, not run on hardware
//
#include <hip/hip_runtime.h>
#include <stdint.h>

typedef _Float16 f16;
typedef __attribute__((ext_vector_type(16))) _Float16 v16h;
typedef __attribute__((ext_vector_type(8)))  _Float16 v8h;
typedef __attribute__((ext_vector_type(8)))  float    v8f;

// ---------------------------------------------------------------- helpers

__device__ __forceinline__ v8f wmma32(v16h a, v16h b, v8f c) {
  // D = A(16x32 f16) x B(32x16 f16) + C(16x16 f32)
  return __builtin_amdgcn_wmma_f32_16x16x32_f16(false, a, false, b, (short)0, c,
                                                false, false);
}

__device__ __forceinline__ v8f splat8(float x) {
  v8f r = {x, x, x, x, x, x, x, x};
  return r;
}

// A-fragment from a row-major [16 x K] f16 tile (ISA 7.12.2 16-bit A layout):
// lane L holds row L%16; K-octet at kbase+8*(L/16) and kbase+16+8*(L/16).
__device__ __forceinline__ v16h load_a(const f16* tile, int K, int kbase, int lane) {
  const int row  = lane & 15;
  const int half = lane >> 4;
  const f16* p = tile + row * K + kbase + half * 8;
  v8h lo = *reinterpret_cast<const v8h*>(p);
  v8h hi = *reinterpret_cast<const v8h*>(p + 16);
  return __builtin_shufflevector(lo, hi, 0, 1, 2, 3, 4, 5, 6, 7,
                                 8, 9, 10, 11, 12, 13, 14, 15);
}

// order-preserving monotone map for float atomic max/min on uint
__device__ __forceinline__ unsigned fmap(float f) {
  unsigned u = __float_as_uint(f);
  return (u & 0x80000000u) ? ~u : (u | 0x80000000u);
}
__device__ __forceinline__ float funmap(unsigned u) {
  unsigned b = (u & 0x80000000u) ? (u ^ 0x80000000u) : ~u;
  return __uint_as_float(b);
}

// ---------------------------------------------------------------- utility kernels

__global__ void fill_u32(unsigned* __restrict__ p, unsigned v, size_t n) {
  size_t i = (size_t)blockIdx.x * blockDim.x + threadIdx.x;
  size_t stride = (size_t)gridDim.x * blockDim.x;
  for (; i < n; i += stride) p[i] = v;
}

// Pack W[K][Nout] (f32 row-major) into per-lane WMMA B fragments:
// frag(kc,nt): lane L holds B[kc*32 + 8*(L/16) + t + (t>=8 ? 8 : 0)][nt*16 + L%16], t=0..15.
__global__ void pack_b(const float* __restrict__ W, f16* __restrict__ out,
                       int K, int Nout) {
  int tid = blockIdx.x * blockDim.x + threadIdx.x;
  int ntiles = Nout >> 4;
  int total = (K >> 5) * ntiles * 32;
  if (tid >= total) return;
  int lane = tid & 31;
  int frag = tid >> 5;
  int nt = frag % ntiles;
  int kc = frag / ntiles;
  int col = nt * 16 + (lane & 15);
  int krow = kc * 32 + 8 * (lane >> 4);
  f16* o = out + (size_t)tid * 16;
#pragma unroll
  for (int t = 0; t < 16; ++t) o[t] = (f16)W[(size_t)(krow + t + (t >= 8 ? 8 : 0)) * Nout + col];
}

__global__ void compute_dinv(const float* __restrict__ cnt, float* __restrict__ dinv, int N) {
  int i = blockIdx.x * blockDim.x + threadIdx.x;
  if (i < N) dinv[i] = rsqrtf(cnt[i] + 1.0f);  // deg includes self-loop, >= 1
}

// ---------------------------------------------------------------- layer 0 edge (scalar f32)

__global__ void l0_edge(const float* __restrict__ x, const int* __restrict__ src,
                        const int* __restrict__ dst, const float* __restrict__ ea,
                        const float* __restrict__ We0, const float* __restrict__ be0,
                        const float* __restrict__ Wpre0, const float* __restrict__ bpre0,
                        float* __restrict__ sum0, float* __restrict__ ssq0,
                        unsigned* __restrict__ mx0, unsigned* __restrict__ mn0,
                        float* __restrict__ cnt, int E) {
  int e = blockIdx.x * blockDim.x + threadIdx.x;
  if (e >= E) return;
  const int s = src[e], d = dst[e];
  const float a0 = ea[(size_t)e * 4 + 0], a1 = ea[(size_t)e * 4 + 1];
  const float a2 = ea[(size_t)e * 4 + 2], a3 = ea[(size_t)e * 4 + 3];
  float ev[16], xd[16], xs[16];
#pragma unroll
  for (int j = 0; j < 16; ++j)
    ev[j] = be0[j] + a0 * We0[j] + a1 * We0[16 + j] + a2 * We0[32 + j] + a3 * We0[48 + j];
#pragma unroll
  for (int j = 0; j < 16; ++j) {
    xd[j] = x[(size_t)d * 16 + j];
    xs[j] = x[(size_t)s * 16 + j];
  }
  for (int f = 0; f < 16; ++f) {
    float m = bpre0[f];
#pragma unroll
    for (int j = 0; j < 16; ++j) m += xd[j] * Wpre0[j * 16 + f];
#pragma unroll
    for (int j = 0; j < 16; ++j) m += xs[j] * Wpre0[(16 + j) * 16 + f];
#pragma unroll
    for (int j = 0; j < 16; ++j) m += ev[j] * Wpre0[(32 + j) * 16 + f];
    const size_t idx = (size_t)d * 16 + f;
    atomicAdd(&sum0[idx], m);
    atomicAdd(&ssq0[idx], m * m);
    atomicMax(&mx0[idx], fmap(m));
    atomicMin(&mn0[idx], fmap(m));
  }
  atomicAdd(&cnt[d], 1.0f);
}

// ---------------------------------------------------------------- layer 0 node (WMMA)
// z[16x96]=[x|sum|mean|max|min|std] -> @Wpost0(96x64) -> @Wlin0(64x64) -> relu -> h1 f16

__global__ void __launch_bounds__(128) l0_node(
    const float* __restrict__ x, const float* __restrict__ sum0,
    const float* __restrict__ ssq0, const unsigned* __restrict__ mx0,
    const unsigned* __restrict__ mn0, const float* __restrict__ cnt,
    const v16h* __restrict__ pWpost0, const float* __restrict__ bpost0,
    const v16h* __restrict__ pWlin0, const float* __restrict__ blin0,
    f16* __restrict__ h1, int N) {
  __shared__ __align__(16) f16 zs[4][16 * 96];
  __shared__ __align__(16) f16 ms[4][16 * 64];
  const int lane = threadIdx.x & 31;
  const int wv = threadIdx.x >> 5;
  const int base = blockIdx.x * 64 + wv * 16;
  const int row = lane & 15;
  const int half = lane >> 4;
  int node = base + row;
  if (node >= N) node = N - 1;
  const float c = cnt[node];
  const float denom = fmaxf(c, 1.0f);
  const bool has = c > 0.0f;
  f16* zr = &zs[wv][row * 96];
  if (half == 0) {
    const float* xp = x + (size_t)node * 16;
    const float* sp = sum0 + (size_t)node * 16;
#pragma unroll
    for (int j = 0; j < 16; ++j) {
      float s = sp[j];
      zr[j]      = (f16)xp[j];
      zr[16 + j] = (f16)s;
      zr[32 + j] = (f16)(s / denom);
    }
  } else {
    const float* sp = sum0 + (size_t)node * 16;
    const float* qp = ssq0 + (size_t)node * 16;
    const unsigned* xp = mx0 + (size_t)node * 16;
    const unsigned* np = mn0 + (size_t)node * 16;
#pragma unroll
    for (int j = 0; j < 16; ++j) {
      float mean = sp[j] / denom;
      float var = fmaxf(qp[j] / denom - mean * mean, 0.0f);
      zr[48 + j] = (f16)(has ? funmap(xp[j]) : 0.0f);
      zr[64 + j] = (f16)(has ? funmap(np[j]) : 0.0f);
      zr[80 + j] = (f16)sqrtf(var + 1e-5f);
    }
  }
  __syncthreads();
  const int col = lane & 15;
  const int rb = half * 8;
  v8f acc[4];
#pragma unroll
  for (int t = 0; t < 4; ++t) acc[t] = splat8(bpost0[t * 16 + col]);
#pragma unroll
  for (int kc = 0; kc < 3; ++kc) {
    v16h a = load_a(zs[wv], 96, kc * 32, lane);
#pragma unroll
    for (int t = 0; t < 4; ++t)
      acc[t] = wmma32(a, pWpost0[(kc * 4 + t) * 32 + lane], acc[t]);
  }
#pragma unroll
  for (int t = 0; t < 4; ++t)
#pragma unroll
    for (int v = 0; v < 8; ++v)
      ms[wv][(rb + v) * 64 + t * 16 + col] = (f16)acc[t][v];
  __syncthreads();
  v8f acc2[4];
#pragma unroll
  for (int t = 0; t < 4; ++t) acc2[t] = splat8(blin0[t * 16 + col]);
#pragma unroll
  for (int kc = 0; kc < 2; ++kc) {
    v16h a = load_a(ms[wv], 64, kc * 32, lane);
#pragma unroll
    for (int t = 0; t < 4; ++t)
      acc2[t] = wmma32(a, pWlin0[(kc * 4 + t) * 32 + lane], acc2[t]);
  }
#pragma unroll
  for (int t = 0; t < 4; ++t)
#pragma unroll
    for (int v = 0; v < 8; ++v) {
      int nd = base + rb + v;
      if (nd < N) h1[(size_t)nd * 64 + t * 16 + col] = (f16)fmaxf(acc2[t][v], 0.0f);
    }
}

// ---------------------------------------------------------------- layer 1 edge (WMMA + atomic scatter)

__global__ void __launch_bounds__(128) l1_edge(
    const f16* __restrict__ h1, const int* __restrict__ src, const int* __restrict__ dst,
    const float* __restrict__ ea, const float* __restrict__ We1, const float* __restrict__ be1,
    const v16h* __restrict__ pWpre1, const float* __restrict__ bpre1,
    float* __restrict__ sum1, float* __restrict__ ssq1,
    unsigned* __restrict__ mx1, unsigned* __restrict__ mn1, int E) {
  __shared__ __align__(16) f16 hs[4][16 * 192];
  const int lane = threadIdx.x & 31;
  const int wv = threadIdx.x >> 5;
  const long base = (long)blockIdx.x * 64 + wv * 16;
  const int row = lane & 15;
  const int half = lane >> 4;
  long e = base + row;
  if (e >= E) e = E - 1;
  const int s = src[e], d = dst[e];
  f16* hr = &hs[wv][row * 192];
  {
    const v8h* pd = reinterpret_cast<const v8h*>(h1 + (size_t)d * 64 + half * 32);
    const v8h* ps = reinterpret_cast<const v8h*>(h1 + (size_t)s * 64 + half * 32);
    v8h* od = reinterpret_cast<v8h*>(hr + half * 32);
    v8h* os = reinterpret_cast<v8h*>(hr + 64 + half * 32);
#pragma unroll
    for (int i = 0; i < 4; ++i) {
      od[i] = pd[i];
      os[i] = ps[i];
    }
  }
  {
    const float a0 = ea[(size_t)e * 4 + 0], a1 = ea[(size_t)e * 4 + 1];
    const float a2 = ea[(size_t)e * 4 + 2], a3 = ea[(size_t)e * 4 + 3];
#pragma unroll
    for (int j = 0; j < 32; ++j) {
      const int cc = half * 32 + j;
      float v = be1[cc] + a0 * We1[cc] + a1 * We1[64 + cc] + a2 * We1[128 + cc] +
                a3 * We1[192 + cc];
      hr[128 + cc] = (f16)v;
    }
  }
  __syncthreads();
  const int col = lane & 15;
  const int rb = half * 8;
  v8f acc[4];
#pragma unroll
  for (int t = 0; t < 4; ++t) acc[t] = splat8(bpre1[t * 16 + col]);
#pragma unroll
  for (int kc = 0; kc < 6; ++kc) {
    v16h a = load_a(hs[wv], 192, kc * 32, lane);
#pragma unroll
    for (int t = 0; t < 4; ++t)
      acc[t] = wmma32(a, pWpre1[(kc * 4 + t) * 32 + lane], acc[t]);
  }
#pragma unroll
  for (int v = 0; v < 8; ++v) {
    const long ed = base + rb + v;
    if (ed < E) {
      const size_t dd = (size_t)dst[ed] * 64;
#pragma unroll
      for (int t = 0; t < 4; ++t) {
        const float val = acc[t][v];
        const size_t idx = dd + t * 16 + col;
        atomicAdd(&sum1[idx], val);
        atomicAdd(&ssq1[idx], val * val);
        atomicMax(&mx1[idx], fmap(val));
        atomicMin(&mn1[idx], fmap(val));
      }
    }
  }
}

// ---------------------------------------------------------------- layer 1 node (WMMA, fused GCN lin)
// z[16x384] -> @Wpost1(384x64) -> @Wlin1(64x64) -> relu -> @Wg(64x16) -> gh f32

__global__ void __launch_bounds__(128) l1_node(
    const f16* __restrict__ h1, const float* __restrict__ sum1,
    const float* __restrict__ ssq1, const unsigned* __restrict__ mx1,
    const unsigned* __restrict__ mn1, const float* __restrict__ cnt,
    const v16h* __restrict__ pWpost1, const float* __restrict__ bpost1,
    const v16h* __restrict__ pWlin1, const float* __restrict__ blin1,
    const v16h* __restrict__ pWg, float* __restrict__ gh, int N) {
  __shared__ __align__(16) f16 zs[4][16 * 384];
  __shared__ __align__(16) f16 ms[4][16 * 64];
  const int lane = threadIdx.x & 31;
  const int wv = threadIdx.x >> 5;
  const int base = blockIdx.x * 64 + wv * 16;
  const int row = lane & 15;
  const int half = lane >> 4;
  int node = base + row;
  if (node >= N) node = N - 1;
  const float c = cnt[node];
  const float denom = fmaxf(c, 1.0f);
  const bool has = c > 0.0f;
  f16* zr = &zs[wv][row * 384];
  {
    const f16* hp = h1 + (size_t)node * 64;
    const float* sp = sum1 + (size_t)node * 64;
    const float* qp = ssq1 + (size_t)node * 64;
    const unsigned* xp = mx1 + (size_t)node * 64;
    const unsigned* np = mn1 + (size_t)node * 64;
#pragma unroll 8
    for (int j = 0; j < 32; ++j) {
      const int cc = half * 32 + j;
      float s = sp[cc];
      float mean = s / denom;
      float var = fmaxf(qp[cc] / denom - mean * mean, 0.0f);
      zr[cc]        = hp[cc];
      zr[64 + cc]   = (f16)s;
      zr[128 + cc]  = (f16)mean;
      zr[192 + cc]  = (f16)(has ? funmap(xp[cc]) : 0.0f);
      zr[256 + cc]  = (f16)(has ? funmap(np[cc]) : 0.0f);
      zr[320 + cc]  = (f16)sqrtf(var + 1e-5f);
    }
  }
  __syncthreads();
  const int col = lane & 15;
  const int rb = half * 8;
  v8f acc[4];
#pragma unroll
  for (int t = 0; t < 4; ++t) acc[t] = splat8(bpost1[t * 16 + col]);
  for (int kc = 0; kc < 12; ++kc) {
    v16h a = load_a(zs[wv], 384, kc * 32, lane);
#pragma unroll
    for (int t = 0; t < 4; ++t)
      acc[t] = wmma32(a, pWpost1[(kc * 4 + t) * 32 + lane], acc[t]);
  }
#pragma unroll
  for (int t = 0; t < 4; ++t)
#pragma unroll
    for (int v = 0; v < 8; ++v)
      ms[wv][(rb + v) * 64 + t * 16 + col] = (f16)acc[t][v];
  __syncthreads();
  v8f acc2[4];
#pragma unroll
  for (int t = 0; t < 4; ++t) acc2[t] = splat8(blin1[t * 16 + col]);
#pragma unroll
  for (int kc = 0; kc < 2; ++kc) {
    v16h a = load_a(ms[wv], 64, kc * 32, lane);
#pragma unroll
    for (int t = 0; t < 4; ++t)
      acc2[t] = wmma32(a, pWlin1[(kc * 4 + t) * 32 + lane], acc2[t]);
  }
  __syncthreads();  // all waves done reading ms before overwrite
#pragma unroll
  for (int t = 0; t < 4; ++t)
#pragma unroll
    for (int v = 0; v < 8; ++v)
      ms[wv][(rb + v) * 64 + t * 16 + col] = (f16)fmaxf(acc2[t][v], 0.0f);
  __syncthreads();
  v8f acc3 = splat8(0.0f);  // GCN lin has no bias before aggregation
#pragma unroll
  for (int kc = 0; kc < 2; ++kc) {
    v16h a = load_a(ms[wv], 64, kc * 32, lane);
    acc3 = wmma32(a, pWg[kc * 32 + lane], acc3);
  }
#pragma unroll
  for (int v = 0; v < 8; ++v) {
    int nd = base + rb + v;
    if (nd < N) gh[(size_t)nd * 16 + col] = acc3[v];
  }
}

// ---------------------------------------------------------------- GCN scatter + head

__global__ void gcn_scatter(const int* __restrict__ src, const int* __restrict__ dst,
                            const float* __restrict__ dinv, const float* __restrict__ gh,
                            float* __restrict__ gcnacc, int E) {
  int e = blockIdx.x * blockDim.x + threadIdx.x;
  if (e >= E) return;
  const int s = src[e], d = dst[e];
  const float wgt = dinv[s] * dinv[d];
  const float* g = gh + (size_t)s * 16;
  float* o = gcnacc + (size_t)d * 16;
#pragma unroll
  for (int c = 0; c < 16; ++c) atomicAdd(&o[c], g[c] * wgt);
}

__global__ void head_kernel(const float* __restrict__ x, const float* __restrict__ gcnacc,
                            const float* __restrict__ gh, const float* __restrict__ dinv,
                            const float* __restrict__ bg, const float* __restrict__ Wh1,
                            const float* __restrict__ bh1, const float* __restrict__ Wh2,
                            const float* __restrict__ bh2, float* __restrict__ out, int N) {
  int i = blockIdx.x * blockDim.x + threadIdx.x;
  if (i >= N) return;
  float t[32];
  const float di = dinv[i];
  const float wl = di * di;  // self-loop weight
#pragma unroll
  for (int c = 0; c < 16; ++c)
    t[c] = gcnacc[(size_t)i * 16 + c] + gh[(size_t)i * 16 + c] * wl + bg[c];
#pragma unroll
  for (int c = 0; c < 16; ++c) t[16 + c] = x[(size_t)i * 16 + c];
  float a[10];
#pragma unroll
  for (int k = 0; k < 10; ++k) {
    float acc = bh1[k];
#pragma unroll
    for (int j = 0; j < 32; ++j) acc += t[j] * Wh1[j * 10 + k];
    a[k] = fmaxf(acc, 0.0f);
  }
#pragma unroll
  for (int k = 0; k < 10; ++k) {
    float acc = bh2[k];
#pragma unroll
    for (int j = 0; j < 10; ++j) acc += a[j] * Wh2[j * 10 + k];
    out[(size_t)i * 10 + k] = acc;
  }
}

// ---------------------------------------------------------------- launch

extern "C" void kernel_launch(void* const* d_in, const int* in_sizes, int n_in,
                              void* d_out, int out_size, void* d_ws, size_t ws_size,
                              hipStream_t stream) {
  const float* x      = (const float*)d_in[0];
  const int*   eidx   = (const int*)d_in[1];
  const float* eattr  = (const float*)d_in[2];
  const float* We0    = (const float*)d_in[3];
  const float* be0    = (const float*)d_in[4];
  const float* Wpre0  = (const float*)d_in[5];
  const float* bpre0  = (const float*)d_in[6];
  const float* Wpost0 = (const float*)d_in[7];
  const float* bpost0 = (const float*)d_in[8];
  const float* Wlin0  = (const float*)d_in[9];
  const float* blin0  = (const float*)d_in[10];
  const float* We1    = (const float*)d_in[11];
  const float* be1    = (const float*)d_in[12];
  const float* Wpre1  = (const float*)d_in[13];
  const float* bpre1  = (const float*)d_in[14];
  const float* Wpost1 = (const float*)d_in[15];
  const float* bpost1 = (const float*)d_in[16];
  const float* Wlin1  = (const float*)d_in[17];
  const float* blin1  = (const float*)d_in[18];
  const float* Wg     = (const float*)d_in[19];
  const float* bg     = (const float*)d_in[20];
  const float* Wh1    = (const float*)d_in[21];
  const float* bh1    = (const float*)d_in[22];
  const float* Wh2    = (const float*)d_in[23];
  const float* bh2    = (const float*)d_in[24];

  const int N = in_sizes[0] / 16;
  const int E = in_sizes[2] / 4;
  const int* src = eidx;
  const int* dst = eidx + E;

  uint8_t* wsp = (uint8_t*)d_ws;
  size_t off = 0;
  auto alloc = [&](size_t bytes) -> void* {
    void* p = wsp + off;
    off += (bytes + 255) & ~(size_t)255;
    return p;
  };
  f16*      h1      = (f16*)alloc((size_t)N * 64 * 2);
  float*    cnt     = (float*)alloc((size_t)N * 4);
  float*    dinv    = (float*)alloc((size_t)N * 4);
  float*    gh      = (float*)alloc((size_t)N * 16 * 4);
  float*    gcnacc  = (float*)alloc((size_t)N * 16 * 4);
  float*    sum0    = (float*)alloc((size_t)N * 16 * 4);
  float*    ssq0    = (float*)alloc((size_t)N * 16 * 4);
  unsigned* mx0     = (unsigned*)alloc((size_t)N * 16 * 4);
  unsigned* mn0     = (unsigned*)alloc((size_t)N * 16 * 4);
  float*    sum1    = (float*)alloc((size_t)N * 64 * 4);
  float*    ssq1    = (float*)alloc((size_t)N * 64 * 4);
  unsigned* mx1     = (unsigned*)alloc((size_t)N * 64 * 4);
  unsigned* mn1     = (unsigned*)alloc((size_t)N * 64 * 4);
  f16* pWpost0 = (f16*)alloc((size_t)3 * 4 * 32 * 16 * 2);
  f16* pWlin0  = (f16*)alloc((size_t)2 * 4 * 32 * 16 * 2);
  f16* pWpre1  = (f16*)alloc((size_t)6 * 4 * 32 * 16 * 2);
  f16* pWpost1 = (f16*)alloc((size_t)12 * 4 * 32 * 16 * 2);
  f16* pWlin1  = (f16*)alloc((size_t)2 * 4 * 32 * 16 * 2);
  f16* pWg     = (f16*)alloc((size_t)2 * 1 * 32 * 16 * 2);

  const unsigned MAPNEG = 0x007FFFFFu;  // fmap(-inf): identity for max
  const unsigned MAPPOS = 0xFF800000u;  // fmap(+inf): identity for min

  fill_u32<<<2048, 256, 0, stream>>>((unsigned*)cnt, 0u, (size_t)N);
  fill_u32<<<2048, 256, 0, stream>>>((unsigned*)sum0, 0u, (size_t)N * 16 * 2);  // sum0+ssq0 contiguous
  fill_u32<<<2048, 256, 0, stream>>>(mx0, MAPNEG, (size_t)N * 16);
  fill_u32<<<2048, 256, 0, stream>>>(mn0, MAPPOS, (size_t)N * 16);
  fill_u32<<<2048, 256, 0, stream>>>((unsigned*)sum1, 0u, (size_t)N * 64 * 2);  // sum1+ssq1 contiguous
  fill_u32<<<2048, 256, 0, stream>>>(mx1, MAPNEG, (size_t)N * 64);
  fill_u32<<<2048, 256, 0, stream>>>(mn1, MAPPOS, (size_t)N * 64);
  fill_u32<<<2048, 256, 0, stream>>>((unsigned*)gcnacc, 0u, (size_t)N * 16);

  auto pack = [&](const float* W, f16* out, int K, int Nout) {
    int total = (K / 32) * (Nout / 16) * 32;
    pack_b<<<(total + 127) / 128, 128, 0, stream>>>(W, out, K, Nout);
  };
  pack(Wpost0, pWpost0, 96, 64);
  pack(Wlin0, pWlin0, 64, 64);
  pack(Wpre1, pWpre1, 192, 64);
  pack(Wpost1, pWpost1, 384, 64);
  pack(Wlin1, pWlin1, 64, 64);
  pack(Wg, pWg, 64, 16);

  l0_edge<<<(E + 255) / 256, 256, 0, stream>>>(x, src, dst, eattr, We0, be0, Wpre0,
                                               bpre0, sum0, ssq0, mx0, mn0, cnt, E);
  compute_dinv<<<(N + 255) / 256, 256, 0, stream>>>(cnt, dinv, N);
  l0_node<<<(N + 63) / 64, 128, 0, stream>>>(x, sum0, ssq0, mx0, mn0, cnt,
                                             (const v16h*)pWpost0, bpost0,
                                             (const v16h*)pWlin0, blin0, h1, N);
  l1_edge<<<(E + 63) / 64, 128, 0, stream>>>(h1, src, dst, eattr, We1, be1,
                                             (const v16h*)pWpre1, bpre1, sum1, ssq1,
                                             mx1, mn1, E);
  l1_node<<<(N + 63) / 64, 128, 0, stream>>>(h1, sum1, ssq1, mx1, mn1, cnt,
                                             (const v16h*)pWpost1, bpost1,
                                             (const v16h*)pWlin1, blin1,
                                             (const v16h*)pWg, gh, N);
  gcn_scatter<<<(E + 255) / 256, 256, 0, stream>>>(src, dst, dinv, gh, gcnacc, E);
  head_kernel<<<(N + 255) / 256, 256, 0, stream>>>(x, gcnacc, gh, dinv, bg, Wh1, bh1,
                                                   Wh2, bh2, (float*)d_out, N);
}
